// DVRPNet_18580028523107
// MI455X (gfx1250) — compile-verified, matrix operand order checked by reading, not run on hardware
//
#include <hip/hip_runtime.h>
#include <math.h>

// ---------------------------------------------------------------------------
// MI455X (gfx1250) implementation. wave32, WMMA 16x16x32 f16 -> f32 accum.
// All operands between matrix ops stay f16 (weights pre-transposed+converted,
// GEMMs write f16 when the consumer is another WMMA). Inner loops are pure
// b128 f16 loads + v_wmma, no per-element cvt.
// ---------------------------------------------------------------------------

typedef __attribute__((ext_vector_type(16))) _Float16 v16h;
typedef __attribute__((ext_vector_type(8)))  _Float16 v8h;
typedef __attribute__((ext_vector_type(8)))  float    v8f;

#define D_MODEL 256
#define NHEAD   8
#define HDIM    32
#define BATCH   32
#define NNODES  512
#define NTOK    513             // depot + 512 nodes
#define NAGENT  16
#define KSTEPS  8
#define TOKROWS (BATCH * NTOK)  // 16416 = 1026 * 16 (exact M tiles)
#define NEGV    (-3.4028234663852886e+38f)

static __device__ __forceinline__ int imin(int a, int b) { return a < b ? a : b; }

// WMMA fragment layout (CDNA5 ISA 7.12.2), wave32:
//   A (16x32 f16): lane L holds row M=L&15; K-base kh=(L>=16)?8:0; v16h elems
//                  [0..7] = K kh..kh+7, [8..15] = K kh+16..kh+23  (contiguous!)
//   B (32x16 f16): mirror with lane column N=L&15, same K striping.
//   C/D (16x16 f32): elem j: rows M=j (lanes 0-15) / M=j+8 (lanes 16-31),
//                    N = lane&15.
static __device__ __forceinline__ v16h frag_ld(const _Float16* p)
{
    v8h lo = *(const v8h*)p;        // K = kh .. kh+7   (16B aligned)
    v8h hi = *(const v8h*)(p + 16); // K = kh+16 .. kh+23
    return __builtin_shufflevector(lo, hi, 0, 1, 2, 3, 4, 5, 6, 7,
                                           8, 9, 10, 11, 12, 13, 14, 15);
}

// ---------------------------------------------------------------------------
// Weight staging: Wt[n*K + k] = (f16) W[k*N + n]   (one-time, ~3.3 MB total)
// ---------------------------------------------------------------------------
__global__ void wt_kernel(const float* __restrict__ W, _Float16* __restrict__ Wt,
                          int K, int N)
{
    int idx = blockIdx.x * blockDim.x + threadIdx.x;
    if (idx >= K * N) return;
    int k = idx / N, n = idx - k * N;
    Wt[(size_t)n * K + k] = (_Float16)W[idx];
}

// ---------------------------------------------------------------------------
// WMMA GEMM: out = act(A[M,K] @ Wt^T + bias); A f16 row-major, Wt f16 [N,K].
// One wave per 16x16 C tile; inner loop = 4x b128 f16 loads + 1 wmma.
// OUT16: write f16 (consumer is another WMMA); else write f32.
// ---------------------------------------------------------------------------
template <int RELU, int OUT16>
__global__ void wmma_gemm_h(const _Float16* __restrict__ A, const _Float16* __restrict__ Wt,
                            const float* __restrict__ bias, float* __restrict__ C,
                            _Float16* __restrict__ C16, int M, int N, int K, int tiles)
{
    int tile = blockIdx.x * (blockDim.x >> 5) + (threadIdx.x >> 5);
    if (tile >= tiles) return;                 // whole wave exits together
    int ntN = N >> 4;
    int mt = tile / ntN;
    int nt = tile - mt * ntN;
    int lane = threadIdx.x & 31;
    int l15  = lane & 15;
    int kh   = (lane >> 4) << 3;               // 0 or 8

    const _Float16* arow = A  + (size_t)(mt * 16 + l15) * K + kh;
    const _Float16* brow = Wt + (size_t)(nt * 16 + l15) * K + kh;

    v8f acc = {};
    for (int k0 = 0; k0 < K; k0 += 32) {
        v16h af = frag_ld(arow + k0);
        v16h bf = frag_ld(brow + k0);
        acc = __builtin_amdgcn_wmma_f32_16x16x32_f16(false, af, false, bf,
                                                     (short)0, acc, false, false);
    }

    int rb = (lane >> 4) << 3;
    float bv = bias ? bias[nt * 16 + l15] : 0.0f;
#pragma unroll
    for (int j = 0; j < 8; ++j) {
        float v = acc[j] + bv;
        if (RELU) v = fmaxf(v, 0.0f);
        size_t idx = (size_t)(mt * 16 + rb + j) * N + nt * 16 + l15;
        if (OUT16) C16[idx] = (_Float16)v;
        else       C[idx] = v;
    }
}

// ---------------------------------------------------------------------------
// Fused attention for one (batch, head, 16-query tile). Q,K,V,O are f16
// [B,513,256], head h = columns [h*32, h*32+32). Stage 1: S = Q@K^T via WMMA
// (scale applied writing LDS); Stage 2: row softmax in LDS (16x513 f32,
// ~33 KB << 320 KB WGP LDS); Stage 3: O = P@V via WMMA. One wave per block.
// ---------------------------------------------------------------------------
__global__ void attention_kernel(const _Float16* __restrict__ Q,
                                 const _Float16* __restrict__ Kt,
                                 const _Float16* __restrict__ V,
                                 _Float16* __restrict__ O)
{
    __shared__ float sc[16][516];
    __shared__ float rowsum[16];

    int b = blockIdx.z, h = blockIdx.y, qt = blockIdx.x;
    int lane = threadIdx.x;
    int l15  = lane & 15;
    int kh   = (lane >> 4) << 3;
    const size_t base = ((size_t)b * NTOK) * D_MODEL + (size_t)h * HDIM;
    const float scale = 0.17677669529663687f;   // 1/sqrt(32)

    // Q fragment (rows clamped for the last partial tile); head dim 32 = one
    // WMMA K-step, so the fragment is loaded once.
    int qrow = imin(qt * 16 + l15, NTOK - 1);
    v16h qa = frag_ld(Q + base + (size_t)qrow * D_MODEL + kh);

    // Stage 1: scores, 33 key tiles of 16
    for (int kt = 0; kt < 33; ++kt) {
        int kcol = kt * 16 + l15;
        v16h kb = frag_ld(Kt + base + (size_t)imin(kcol, NTOK - 1) * D_MODEL + kh);
        v8f s = {};
        s = __builtin_amdgcn_wmma_f32_16x16x32_f16(false, qa, false, kb,
                                                   (short)0, s, false, false);
        int rb = (lane >> 4) << 3;
        if (kcol < NTOK) {
#pragma unroll
            for (int j = 0; j < 8; ++j) sc[rb + j][kcol] = s[j] * scale;
        }
    }
    __syncthreads();

    // Stage 2: softmax (one lane per query row; un-normalized exp kept in LDS)
    if (lane < 16) {
        float m = -INFINITY;
        for (int c = 0; c < NTOK; ++c) m = fmaxf(m, sc[lane][c]);
        float ssum = 0.0f;
        for (int c = 0; c < NTOK; ++c) {
            float e = expf(sc[lane][c] - m);
            sc[lane][c] = e;
            ssum += e;
        }
        rowsum[lane] = ssum;
    }
    __syncthreads();

    // Stage 3: O = P @ V, 17 key tiles of 32 (tail keys contribute P = 0)
    v8f acc0 = {}, acc1 = {};
    for (int kt2 = 0; kt2 < 17; ++kt2) {
        int kb0 = kt2 * 32 + kh;
        v16h pa, vb0, vb1;
#pragma unroll
        for (int j = 0; j < 4; ++j) {
            int c0 = kb0 + 2*j, c1 = kb0 + 2*j + 1;
            int c2 = kb0 + 16 + 2*j, c3 = kb0 + 16 + 2*j + 1;
            pa[2*j]     = (_Float16)((c0 < NTOK) ? sc[l15][c0] : 0.0f);
            pa[2*j+1]   = (_Float16)((c1 < NTOK) ? sc[l15][c1] : 0.0f);
            pa[8+2*j]   = (_Float16)((c2 < NTOK) ? sc[l15][c2] : 0.0f);
            pa[8+2*j+1] = (_Float16)((c3 < NTOK) ? sc[l15][c3] : 0.0f);
            const _Float16* v0 = V + base + (size_t)imin(c0, NTOK - 1) * D_MODEL;
            const _Float16* v1 = V + base + (size_t)imin(c1, NTOK - 1) * D_MODEL;
            const _Float16* v2 = V + base + (size_t)imin(c2, NTOK - 1) * D_MODEL;
            const _Float16* v3 = V + base + (size_t)imin(c3, NTOK - 1) * D_MODEL;
            vb0[2*j]     = v0[l15];
            vb0[2*j+1]   = v1[l15];
            vb0[8+2*j]   = v2[l15];
            vb0[8+2*j+1] = v3[l15];
            vb1[2*j]     = v0[16 + l15];
            vb1[2*j+1]   = v1[16 + l15];
            vb1[8+2*j]   = v2[16 + l15];
            vb1[8+2*j+1] = v3[16 + l15];
        }
        acc0 = __builtin_amdgcn_wmma_f32_16x16x32_f16(false, pa, false, vb0,
                                                      (short)0, acc0, false, false);
        acc1 = __builtin_amdgcn_wmma_f32_16x16x32_f16(false, pa, false, vb1,
                                                      (short)0, acc1, false, false);
    }

    int rb = (lane >> 4) << 3;
#pragma unroll
    for (int j = 0; j < 8; ++j) {
        int q = qt * 16 + rb + j;
        if (q < NTOK) {
            float inv = 1.0f / rowsum[rb + j];
            O[base + (size_t)q * D_MODEL + l15]      = (_Float16)(acc0[j] * inv);
            O[base + (size_t)q * D_MODEL + 16 + l15] = (_Float16)(acc1[j] * inv);
        }
    }
}

// ---------------------------------------------------------------------------
// Residual add + LayerNorm: h = LN(h + x) * g + b (f32, in place), and the
// f16 shadow copy hh for the next WMMA consumer. One wave per row.
// ---------------------------------------------------------------------------
__global__ void resid_ln(float* __restrict__ h, _Float16* __restrict__ hh,
                         const float* __restrict__ x,
                         const float* __restrict__ g, const float* __restrict__ b)
{
    int row  = blockIdx.x * (blockDim.x >> 5) + (threadIdx.x >> 5);
    int lane = threadIdx.x & 31;
    float* hr = h + (size_t)row * D_MODEL;
    _Float16* hhr = hh + (size_t)row * D_MODEL;
    const float* xr = x + (size_t)row * D_MODEL;

    float vals[8];
    float s = 0.0f;
#pragma unroll
    for (int i = 0; i < 8; ++i) {
        vals[i] = hr[lane + 32 * i] + xr[lane + 32 * i];
        s += vals[i];
    }
#pragma unroll
    for (int off = 16; off > 0; off >>= 1) s += __shfl_xor(s, off, 32);
    float mu = s * (1.0f / 256.0f);
    float var = 0.0f;
#pragma unroll
    for (int i = 0; i < 8; ++i) { float d = vals[i] - mu; var += d * d; }
#pragma unroll
    for (int off = 16; off > 0; off >>= 1) var += __shfl_xor(var, off, 32);
    var *= (1.0f / 256.0f);
    float inv = rsqrtf(var + 1e-5f);
#pragma unroll
    for (int i = 0; i < 8; ++i) {
        float o = (vals[i] - mu) * inv * g[lane + 32 * i] + b[lane + 32 * i];
        hr[lane + 32 * i]  = o;
        hhr[lane + 32 * i] = (_Float16)o;
    }
}

// ---------------------------------------------------------------------------
// Token embedding: writes f32 residual stream h and f16 shadow hh.
// ---------------------------------------------------------------------------
__global__ void embed_kernel(const float* __restrict__ nodes, const float* __restrict__ depot,
                             const float* __restrict__ node_w, const float* __restrict__ node_b,
                             const float* __restrict__ depot_w, const float* __restrict__ depot_b,
                             float* __restrict__ h, _Float16* __restrict__ hh)
{
    int tok = blockIdx.x;
    int b = tok / NTOK, t = tok - b * NTOK;
    int c = threadIdx.x;
    float acc;
    if (t == 0) {
        const float* dp = depot + (size_t)b * 3;
        acc = depot_b[c];
#pragma unroll
        for (int i = 0; i < 3; ++i) acc += dp[i] * depot_w[i * D_MODEL + c];
    } else {
        const float* np = nodes + ((size_t)b * NNODES + (t - 1)) * 5;
        acc = node_b[c];
#pragma unroll
        for (int i = 0; i < 5; ++i) acc += np[i] * node_w[i * D_MODEL + c];
    }
    h[(size_t)tok * D_MODEL + c]  = acc;
    hh[(size_t)tok * D_MODEL + c] = (_Float16)acc;
}

// ---------------------------------------------------------------------------
// Agent embedding (f16 out): out[B*A,256] = ag[B*A,4] @ w[4,256] + b
// ---------------------------------------------------------------------------
__global__ void agent_embed(const float* __restrict__ ag, const float* __restrict__ w,
                            const float* __restrict__ bias, _Float16* __restrict__ out)
{
    int row = blockIdx.x;
    int c = threadIdx.x;
    const float* a = ag + (size_t)row * 4;
    float acc = bias[c];
#pragma unroll
    for (int i = 0; i < 4; ++i) acc += a[i] * w[i * D_MODEL + c];
    out[(size_t)row * D_MODEL + c] = (_Float16)acc;
}

// ---------------------------------------------------------------------------
// Pointer scores per batch: raw[b,a,t] = qenc[b*16+a,:] . kmat[b*513+t,:]
// One wave per (batch, 16-token tile); f16 operands, contiguous b128 loads
// on both sides (B-operand is Kmat^T so its K runs along the row).
// ---------------------------------------------------------------------------
__global__ void score_gemm(const _Float16* __restrict__ Qe, const _Float16* __restrict__ Km,
                           float* __restrict__ S)
{
    int b = blockIdx.y, nt = blockIdx.x;
    int lane = threadIdx.x;
    int l15  = lane & 15;
    int kh   = (lane >> 4) << 3;
    int tcol = nt * 16 + l15;

    const _Float16* qrow = Qe + (size_t)(b * NAGENT + l15) * D_MODEL + kh;
    const _Float16* krow = Km + ((size_t)b * NTOK + imin(tcol, NTOK - 1)) * D_MODEL + kh;

    v8f acc = {};
    for (int k0 = 0; k0 < D_MODEL; k0 += 32) {
        v16h af = frag_ld(qrow + k0);
        v16h bf = frag_ld(krow + k0);
        acc = __builtin_amdgcn_wmma_f32_16x16x32_f16(false, af, false, bf,
                                                     (short)0, acc, false, false);
    }
    int rb = (lane >> 4) << 3;
    if (tcol < NTOK) {
#pragma unroll
        for (int j = 0; j < 8; ++j)
            S[((size_t)b * NAGENT + rb + j) * NTOK + tcol] = acc[j];
    }
}

// ---------------------------------------------------------------------------
// Logits post-processing: tanh clip, node mask, lateness penalty, capacity.
// ---------------------------------------------------------------------------
__global__ void logits_post(const float* __restrict__ raw, const unsigned char* __restrict__ mask,
                            const float* __restrict__ ag, const float* __restrict__ nodes,
                            float* __restrict__ logits)
{
    int idx = blockIdx.x * blockDim.x + threadIdx.x;
    if (idx >= BATCH * NAGENT * NTOK) return;
    int b = idx / (NAGENT * NTOK);
    int r = idx - b * (NAGENT * NTOK);
    int a = r / NTOK;
    int t = r - a * NTOK;

    float x = 10.0f * tanhf(raw[idx] * 0.0625f);   // TANH_CLIP * tanh(./sqrt(256))
    if (t >= 1) {
        int n = t - 1;
        const float* nd = nodes + ((size_t)b * NNODES + n) * 5;
        const float* ap = ag + ((size_t)b * NAGENT + a) * 4;
        if (mask[b * NNODES + n]) x = NEGV;
        float dist = fabsf(ap[0] - nd[0]) + fabsf(ap[1] - nd[1]);
        float late = fmaxf(ap[3] + dist - nd[4], 0.0f);
        x -= 0.1f * late;
        if (nd[3] > ap[2]) x = NEGV;               // demand > remaining capacity
    }
    logits[idx] = x;
}

// ---------------------------------------------------------------------------
// Copy mutable state (agents, node mask) from inputs into workspace.
// ---------------------------------------------------------------------------
__global__ void init_state(const float* __restrict__ agents_in,
                           const unsigned char* __restrict__ mask_in,
                           float* __restrict__ ag, unsigned char* __restrict__ mask)
{
    int i = blockIdx.x * blockDim.x + threadIdx.x;
    if (i < BATCH * NAGENT * 4) ag[i] = agents_in[i];
    if (i < BATCH * NNODES)     mask[i] = mask_in[i] ? 1 : 0;
}

// ---------------------------------------------------------------------------
// Greedy assignment (16 sequential rounds, matching jnp.argmax first-max ties)
// + agent/mask state update + output write. One wave per batch.
// ---------------------------------------------------------------------------
__global__ void greedy_update(const float* __restrict__ logits, unsigned char* __restrict__ mask,
                              float* __restrict__ ag, const float* __restrict__ nodes,
                              const float* __restrict__ depot, const float* __restrict__ capf,
                              float* __restrict__ out, int step)
{
    int b = blockIdx.x;
    int lane = threadIdx.x;
    __shared__ unsigned char used[NTOK];
    __shared__ unsigned char assigned[NAGENT];
    __shared__ int   sel[NAGENT];
    __shared__ float bestScore[NAGENT];
    __shared__ int   bestCand[NAGENT];

    for (int i = lane; i < NTOK; i += 32) used[i] = 0;
    if (lane < NAGENT) { assigned[lane] = 0; sel[lane] = 0; }
    __syncthreads();

    for (int round = 0; round < NAGENT; ++round) {
        if (lane < NAGENT) {
            float bs = -INFINITY;
            int bc = 0;
            if (!assigned[lane]) {
                const float* lr = logits + ((size_t)b * NAGENT + lane) * NTOK;
                for (int c = 0; c < NTOK; ++c) {
                    if (used[c]) continue;
                    float v = lr[c];
                    if (v > bs) { bs = v; bc = c; }   // strict > => first max
                }
            }
            bestScore[lane] = bs;
            bestCand[lane]  = bc;
        }
        __syncthreads();
        if (lane == 0) {
            float bs = -INFINITY;
            int ba = 0;
            for (int a = 0; a < NAGENT; ++a)
                if (bestScore[a] > bs) { bs = bestScore[a]; ba = a; }
            int bi = bestCand[ba];
            sel[ba] = bi;
            assigned[ba] = 1;
            if (bi >= 1) used[bi] = 1;                // depot never marked used
        }
        __syncthreads();
    }

    if (lane < NAGENT) {
        int a = lane;
        int s = sel[a];
        float* ap = ag + ((size_t)b * NAGENT + a) * 4;
        int ni = (s - 1 > 0) ? (s - 1) : 0;
        const float* nd = nodes + ((size_t)b * NNODES + ni) * 5;
        int dx, dy;
        if (s >= 1) { dx = (int)nd[0]; dy = (int)nd[1]; }
        else        { dx = (int)depot[(size_t)b * 3 + 0]; dy = (int)depot[(size_t)b * 3 + 1]; }
        int cx = (int)ap[0], cy = (int)ap[1];
        int adx = cx > dx ? cx - dx : dx - cx;
        int ady = cy > dy ? cy - dy : dy - cy;
        float dist = (float)(adx + ady);
        float new_s = (s >= 1) ? fmaxf(ap[2] - nd[3], 0.0f) : capf[b * NAGENT + a];
        ap[0] = (float)dx; ap[1] = (float)dy; ap[2] = new_s; ap[3] = ap[3] + dist;

        out[((size_t)b * NAGENT + a) * KSTEPS + step] = (float)s;
        float* oxy = out + BATCH * NAGENT * KSTEPS
                   + (((size_t)b * NAGENT + a) * KSTEPS + step) * 2;
        oxy[0] = (float)dx; oxy[1] = (float)dy;
    }
    __syncthreads();
    for (int i = lane; i < NNODES; i += 32) mask[b * NNODES + i] |= used[i + 1];
}

// ---------------------------------------------------------------------------
// Host orchestration. Input order = flattened setup_inputs() dict order:
//  0 nodes  1 node_mask  2 depot  3 agents  4 cap_full  5 k
//  6 node_w 7 node_b 8 depot_w 9 depot_b 10 agent_w 11 agent_b
//  12+12*L: wq wk wv wo ln1_g ln1_b ln2_g ln2_b ffn_w1 ffn_b1 ffn_w2 ffn_b2
//  36 dec_wq 37 dec_wk
// ---------------------------------------------------------------------------
extern "C" void kernel_launch(void* const* d_in, const int* in_sizes, int n_in,
                              void* d_out, int out_size, void* d_ws, size_t ws_size,
                              hipStream_t stream)
{
    (void)in_sizes; (void)n_in; (void)out_size; (void)ws_size;

    const float* nodes   = (const float*)d_in[0];
    const unsigned char* node_mask = (const unsigned char*)d_in[1];
    const float* depot   = (const float*)d_in[2];
    const float* agents  = (const float*)d_in[3];
    const float* capf    = (const float*)d_in[4];
    const float* node_w  = (const float*)d_in[6];
    const float* node_b  = (const float*)d_in[7];
    const float* depot_w = (const float*)d_in[8];
    const float* depot_b = (const float*)d_in[9];
    const float* agent_w = (const float*)d_in[10];
    const float* agent_b = (const float*)d_in[11];
    const float* dec_wq  = (const float*)d_in[36];
    const float* dec_wk  = (const float*)d_in[37];

    // Workspace layout:
    //  h     f32 [16416,256]  residual stream                     (16.8 MB)
    //  hh    f16 [16416,256]  shadow of h for WMMA A operands     ( 8.4 MB)
    //  big16 f16 [16416,1024] q/k/v/attnO slices, FFN hidden,
    //                         then all decoder scratch            (33.6 MB)
    //  tmp   f32 [16416,256]  proj/FFN output; later Kmat (f16)   (16.8 MB)
    //  wstage f16             transposed weights                  ( 3.3 MB)
    float*    h     = (float*)d_ws;
    _Float16* hh    = (_Float16*)(h + (size_t)TOKROWS * D_MODEL);
    _Float16* big16 = hh + (size_t)TOKROWS * D_MODEL;
    float*    tmp   = (float*)(big16 + (size_t)TOKROWS * 1024);
    _Float16* wstage = (_Float16*)(tmp + (size_t)TOKROWS * D_MODEL);

    _Float16* qb16 = big16;
    _Float16* kb16 = big16 + 1 * (size_t)TOKROWS * D_MODEL;
    _Float16* vb16 = big16 + 2 * (size_t)TOKROWS * D_MODEL;
    _Float16* ob16 = big16 + 3 * (size_t)TOKROWS * D_MODEL;
    _Float16* hid16 = big16;                     // FFN hidden [16416,1024]
    _Float16* kmat16 = (_Float16*)tmp;           // decoder Kmat (f16 view)

    _Float16* agemb16 = big16;                                   // 512*256
    _Float16* qenc16  = agemb16 + (size_t)BATCH * NAGENT * D_MODEL;
    float* raw    = (float*)(qenc16 + (size_t)BATCH * NAGENT * D_MODEL);
    float* logits = raw + (size_t)BATCH * NAGENT * NTOK;
    float* ag     = logits + (size_t)BATCH * NAGENT * NTOK;
    unsigned char* maskst = (unsigned char*)(ag + (size_t)BATCH * NAGENT * 4);

    // Transposed-f16 weight staging offsets (halves)
    const size_t PERL = 4 * 65536 + 2 * 262144;  // per-layer: 4 attn + 2 ffn
    auto wtp = [&](int L, int which) -> _Float16* {
        static const size_t off[6] = {0, 65536, 131072, 196608, 262144, 524288};
        return wstage + (size_t)L * PERL + off[which];
    };
    _Float16* dec_wqT = wstage + 2 * PERL;
    _Float16* dec_wkT = dec_wqT + 65536;

    auto stage_wt = [&](const float* W, _Float16* Wt, int K, int N) {
        int n = K * N;
        wt_kernel<<<(n + 255) / 256, 256, 0, stream>>>(W, Wt, K, N);
    };
    auto gemm = [&](const _Float16* A, const _Float16* Wt, const float* bias,
                    float* C, _Float16* C16, int M, int N, int K, int relu, int out16) {
        int tiles = (M / 16) * (N / 16);
        int blocks = (tiles + 7) / 8;            // 8 waves per 256-thread block
        if (out16) {
            if (relu) wmma_gemm_h<1, 1><<<blocks, 256, 0, stream>>>(A, Wt, bias, C, C16, M, N, K, tiles);
            else      wmma_gemm_h<0, 1><<<blocks, 256, 0, stream>>>(A, Wt, bias, C, C16, M, N, K, tiles);
        } else {
            wmma_gemm_h<0, 0><<<blocks, 256, 0, stream>>>(A, Wt, bias, C, C16, M, N, K, tiles);
        }
    };

    // ---- Stage weights (one-time per launch) ----
    for (int L = 0; L < 2; ++L) {
        for (int w = 0; w < 4; ++w)               // wq wk wv wo: [256,256]
            stage_wt((const float*)d_in[12 + 12 * L + w], wtp(L, w), D_MODEL, D_MODEL);
        stage_wt((const float*)d_in[12 + 12 * L + 8],  wtp(L, 4), D_MODEL, 1024);  // ffn_w1
        stage_wt((const float*)d_in[12 + 12 * L + 10], wtp(L, 5), 1024, D_MODEL);  // ffn_w2
    }
    stage_wt(dec_wq, dec_wqT, D_MODEL, D_MODEL);
    stage_wt(dec_wk, dec_wkT, D_MODEL, D_MODEL);

    // ---- Encoder ----
    embed_kernel<<<TOKROWS, D_MODEL, 0, stream>>>(nodes, depot, node_w, node_b,
                                                  depot_w, depot_b, h, hh);
    for (int L = 0; L < 2; ++L) {
        const float* ln1g = (const float*)d_in[12 + 12 * L + 4];
        const float* ln1b = (const float*)d_in[12 + 12 * L + 5];
        const float* ln2g = (const float*)d_in[12 + 12 * L + 6];
        const float* ln2b = (const float*)d_in[12 + 12 * L + 7];
        const float* b1   = (const float*)d_in[12 + 12 * L + 9];
        const float* b2   = (const float*)d_in[12 + 12 * L + 11];

        gemm(hh, wtp(L, 0), nullptr, nullptr, qb16, TOKROWS, D_MODEL, D_MODEL, 0, 1);
        gemm(hh, wtp(L, 1), nullptr, nullptr, kb16, TOKROWS, D_MODEL, D_MODEL, 0, 1);
        gemm(hh, wtp(L, 2), nullptr, nullptr, vb16, TOKROWS, D_MODEL, D_MODEL, 0, 1);
        attention_kernel<<<dim3(33, NHEAD, BATCH), 32, 0, stream>>>(qb16, kb16, vb16, ob16);
        gemm(ob16, wtp(L, 3), nullptr, tmp, nullptr, TOKROWS, D_MODEL, D_MODEL, 0, 0);
        resid_ln<<<TOKROWS / 8, 256, 0, stream>>>(h, hh, tmp, ln1g, ln1b);
        gemm(hh, wtp(L, 4), b1, nullptr, hid16, TOKROWS, 1024, D_MODEL, 1, 1);   // relu
        gemm(hid16, wtp(L, 5), b2, tmp, nullptr, TOKROWS, D_MODEL, 1024, 0, 0);
        resid_ln<<<TOKROWS / 8, 256, 0, stream>>>(h, hh, tmp, ln2g, ln2b);
    }

    // ---- Decoder ----
    gemm(hh, dec_wkT, nullptr, nullptr, kmat16, TOKROWS, D_MODEL, D_MODEL, 0, 1);
    init_state<<<64, 256, 0, stream>>>(agents, node_mask, ag, maskst);

    for (int s = 0; s < KSTEPS; ++s) {
        agent_embed<<<BATCH * NAGENT, D_MODEL, 0, stream>>>(ag, agent_w, agent_b, agemb16);
        gemm(agemb16, dec_wqT, nullptr, nullptr, qenc16, BATCH * NAGENT, D_MODEL, D_MODEL, 0, 1);
        score_gemm<<<dim3(33, BATCH), 32, 0, stream>>>(qenc16, kmat16, raw);
        int tot = BATCH * NAGENT * NTOK;
        logits_post<<<(tot + 255) / 256, 256, 0, stream>>>(raw, maskst, ag, nodes, logits);
        greedy_update<<<BATCH, 32, 0, stream>>>(logits, maskst, ag, nodes, depot, capf,
                                                (float*)d_out, s);
    }
}
